// NTXenLoss_NearestNeighbours_23742579213068
// MI455X (gfx1250) — matrix operand order repaired and checked by reading, not compiled, not run on hardware
//
#include <hip/hip_runtime.h>
#include <stdint.h>
#include <math.h>

#define DIM 256
#define LDS_PITCH 132          // dwords per LDS A-tile row (128 + 4 pad -> conflict-free)
#define INV_T 10.0f
#define LSE_C 10.0f            // fixed logsumexp shift: |dot| <= 1 so |logit| <= ~10
#define NEG_INF (-1.0e30f)

typedef __attribute__((ext_vector_type(16))) _Float16 v16h;
typedef __attribute__((ext_vector_type(8)))  float    v8f;
typedef __attribute__((ext_vector_type(4)))  unsigned int u32x4;
typedef __attribute__((ext_vector_type(4)))  int          i32x4;
typedef __attribute__((ext_vector_type(8)))  int          i32x8;

union Frag16 { unsigned int u[8]; uint4 q[2]; v16h v; };

// ---------------------------------------------------------------------------
// Stage the 16x256 f16 A-tile into LDS with the Tensor Data Mover.
// TDM hardware padding (4 DWORDs after every 128 DWORDs = one 512B row)
// reproduces the bank-conflict-free LDS_PITCH=132 layout in hardware.
// Issued by wave 0 only (TDM ignores EXEC; one DMA per block), tracked by
// TENSORcnt. ISA 8.3/8.4 descriptor layout.
// ---------------------------------------------------------------------------
__device__ __forceinline__ void stage_a_tile_tdm(const _Float16* __restrict__ A, int rowbase,
                                                 _Float16* lds, int N) {
  const unsigned long long ga =
      (unsigned long long)(uintptr_t)(A + (size_t)rowbase * DIM);
  const unsigned int la = (unsigned int)(uintptr_t)lds;   // flat addr low 32b == LDS offset

  u32x4 g0;
  g0[0] = 1u;                                             // count=1, user descriptor
  g0[1] = la;                                             // lds_addr (bytes)
  g0[2] = (unsigned int)ga;                               // global_addr[31:0]
  g0[3] = (unsigned int)((ga >> 32) & 0x01FFFFFFull)      // global_addr[56:32]
        | (2u << 30);                                     // type=2 ("image")

  i32x8 g1;
  g1[0] = (int)((1u << 16)        // data_size = 1 -> 2-byte elements
              | (1u << 20)        // pad_enable
              | (6u << 22)        // pad_interval: 128 DWORDs (one 512B row)
              | (3u << 25));      // pad_amount: 4 DWORDs
  g1[1] = (int)(((unsigned)DIM & 0xFFFFu) << 16);               // tensor_dim0[15:0]
  g1[2] = (int)((((unsigned)DIM >> 16) & 0xFFFFu)               // tensor_dim0[31:16]
              | (((unsigned)N & 0xFFFFu) << 16));               // tensor_dim1[15:0]
  g1[3] = (int)((((unsigned)N >> 16) & 0xFFFFu)                 // tensor_dim1[31:16]
              | (((unsigned)DIM & 0xFFFFu) << 16));             // tile_dim0 = 256 elems
  g1[4] = (int)16;                                              // tile_dim1 = 16 rows
  g1[5] = (int)DIM;                                             // tensor_dim0_stride (elems)
  g1[6] = 0;
  g1[7] = 0;

  const i32x4 z4 = (i32x4)0;
#if defined(__clang_major__) && (__clang_major__ >= 23)
  const i32x8 z8 = (i32x8)0;
  __builtin_amdgcn_tensor_load_to_lds(g0, g1, z4, z4, z8, 0);
#else
  __builtin_amdgcn_tensor_load_to_lds(g0, g1, z4, z4, 0);
#endif
}

// ---------------------------------------------------------------------------
// Build the 8 WMMA A-fragments from LDS (ISA 7.12.2 16-bit A 16x32 layout).
// ---------------------------------------------------------------------------
__device__ __forceinline__ void load_a_frags(const unsigned int* lds, int m, int hi,
                                             Frag16 af[8]) {
#pragma unroll
  for (int ks = 0; ks < 8; ++ks) {
    // VGPR v in 0..3: k = ks*32 + hi*8 + 2v(+1);  v in 4..7: k = ks*32 + 16 + hi*8 + ...
    const int kb = ks * 32 + hi * 8;
    const unsigned int* p0 = lds + m * LDS_PITCH + (kb >> 1);
    const unsigned int* p1 = lds + m * LDS_PITCH + ((kb + 16) >> 1);
#pragma unroll
    for (int t = 0; t < 4; ++t) { af[ks].u[t] = p0[t]; af[ks].u[4 + t] = p1[t]; }
  }
}

__device__ __forceinline__ v8f tile_gemm(const Frag16 af[8], const _Float16* __restrict__ B,
                                         int ncol, int hi) {
  const uint4* brow = (const uint4*)(B + (size_t)ncol * DIM);
  // Issue all 16 b128 loads first so they clause and overlap the MMAs.
  Frag16 bf[8];
#pragma unroll
  for (int ks = 0; ks < 8; ++ks) {
    bf[ks].q[0] = brow[ks * 4 + hi];
    bf[ks].q[1] = brow[ks * 4 + 2 + hi];
  }
  v8f acc = {};
#pragma unroll
  for (int ks = 0; ks < 8; ++ks)
    acc = __builtin_amdgcn_wmma_f32_16x16x32_f16(false, af[ks].v, false, bf[ks].v,
                                                 (short)0, acc, false, false);
  return acc;
}

// ---------------------------------------------------------------------------
// 1) L2-normalize rows, emit f16
// ---------------------------------------------------------------------------
__global__ __launch_bounds__(DIM)
void normalize_f16_kernel(const float* __restrict__ z, _Float16* __restrict__ out) {
  __shared__ float red[8];
  const int row = blockIdx.x;
  const int tid = threadIdx.x;                   // DIM == 256 threads
  float x = z[(size_t)row * DIM + tid];
  float ss = x * x;
#pragma unroll
  for (int m = 16; m > 0; m >>= 1) ss += __shfl_xor(ss, m, 32);
  if ((tid & 31) == 0) red[tid >> 5] = ss;
  __syncthreads();
  float tot = 0.f;
#pragma unroll
  for (int i = 0; i < 8; ++i) tot += red[i];
  const float n = fmaxf(sqrtf(tot), 1e-12f);
  out[(size_t)row * DIM + tid] = (_Float16)(x / n);
}

// ---------------------------------------------------------------------------
// 2) Fused GEMM (A @ B^T) + per-row max/argmax, diagonal excluded.
//    Per-lane streaming max; single cross-lane butterfly at the end.
// ---------------------------------------------------------------------------
__global__ __launch_bounds__(256)
void rowmax_kernel(const _Float16* __restrict__ A, const _Float16* __restrict__ B,
                   float* __restrict__ out_val, int* __restrict__ out_idx, int N) {
  __shared__ _Float16 lds_a[16 * LDS_PITCH * 2];
  __shared__ float s_mv[8][16];
  __shared__ int   s_mi[8][16];

  const int tid = threadIdx.x;
  const int lane = tid & 31, wv = tid >> 5;
  const int m = lane & 15, hi = lane >> 4;
  const int rowbase = blockIdx.x * 16;

  if (tid < 32) {                                // wave 0: async DMA the A tile
    stage_a_tile_tdm(A, rowbase, lds_a, N);
    __builtin_amdgcn_s_wait_tensorcnt(0);
  }
  __syncthreads();

  Frag16 af[8];
  load_a_frags((const unsigned int*)lds_a, m, hi, af);

  int growv[8];
  float rmax[8];
  int   ridx[8];
#pragma unroll
  for (int r = 0; r < 8; ++r) {
    growv[r] = rowbase + r + hi * 8;
    rmax[r] = NEG_INF;
    ridx[r] = 0;
  }

  const int ntiles = N / 16;
  for (int nt = wv; nt < ntiles; nt += 8) {
    const int c = nt * 16 + m;                   // this lane's column
    if (nt + 8 < ntiles)                         // global_prefetch of next tile's B row
      __builtin_prefetch(B + (size_t)(c + 128) * DIM, 0, 1);
    v8f acc = tile_gemm(af, B, c, hi);
#pragma unroll
    for (int r = 0; r < 8; ++r) {
      float v = (c == growv[r]) ? NEG_INF : acc[r];   // - INF * eye
      if (v > rmax[r]) { rmax[r] = v; ridx[r] = c; }  // lane cols increase -> first idx
    }
  }

  // cross-lane butterfly argmax (once), first-index tie-break
#pragma unroll
  for (int r = 0; r < 8; ++r) {
    float v = rmax[r]; int c = ridx[r];
#pragma unroll
    for (int mask = 1; mask < 16; mask <<= 1) {
      float ov = __shfl_xor(v, mask, 32);
      int   oc = __shfl_xor(c, mask, 32);
      if (ov > v || (ov == v && oc < c)) { v = ov; c = oc; }
    }
    if (m == 0) { s_mv[wv][r + hi * 8] = v; s_mi[wv][r + hi * 8] = c; }
  }
  __syncthreads();

  if (tid < 16) {
    float bv = NEG_INF; int bi = 0;
    for (int w = 0; w < 8; ++w) {
      float v = s_mv[w][tid]; int c = s_mi[w][tid];
      if (v > bv || (v == bv && c < bi)) { bv = v; bi = c; }
    }
    out_val[rowbase + tid] = bv;
    out_idx[rowbase + tid] = bi;
  }
}

// ---------------------------------------------------------------------------
// 3) Pick nearest neighbour source, gather z_nn rows (f16)
// ---------------------------------------------------------------------------
__global__ __launch_bounds__(128)
void select_gather_kernel(const float* __restrict__ mv_aa, const int* __restrict__ mi_aa,
                          const float* __restrict__ mv_ab, const int* __restrict__ mi_ab,
                          const _Float16* __restrict__ za, const _Float16* __restrict__ zb,
                          _Float16* __restrict__ znn, int* __restrict__ nn_idx,
                          int* __restrict__ nn_src) {
  const int i = blockIdx.x;
  const int tid = threadIdx.x;                   // 128 dwords per row
  const bool use_aa = mv_aa[i] > mv_ab[i];
  const int idx = use_aa ? mi_aa[i] : mi_ab[i];
  const _Float16* src = (use_aa ? za : zb) + (size_t)idx * DIM;
  ((unsigned int*)(znn + (size_t)i * DIM))[tid] = ((const unsigned int*)src)[tid];
  if (tid == 0) { nn_idx[i] = idx; nn_src[i] = use_aa ? 0 : 1; }
}

// ---------------------------------------------------------------------------
// 4) Fused GEMM (Znn @ B^T)/T + masking + fixed-shift sum-exp partials.
//    Logits are bounded (|dot|<=1, /T=10) so exp(v - LSE_C) cannot overflow:
//    no online max needed, inner loop stays branch- and LDS-free.
// ---------------------------------------------------------------------------
__global__ __launch_bounds__(256)
void lse_kernel(const _Float16* __restrict__ A, const _Float16* __restrict__ B,
                const int* __restrict__ nn_idx, const int* __restrict__ nn_src,
                int mask_src_val, int mask_diag,
                float* __restrict__ m_out, float* __restrict__ s_out,
                float* __restrict__ dvec, int N) {
  __shared__ _Float16 lds_a[16 * LDS_PITCH * 2];
  __shared__ float s_s[8][16];
  __shared__ float s_dw[8][16];
  __shared__ int   s_nni[16];
  __shared__ int   s_nns[16];

  const int tid = threadIdx.x;
  const int lane = tid & 31, wv = tid >> 5;
  const int m = lane & 15, hi = lane >> 4;
  const int rowbase = blockIdx.x * 16;

  if (tid < 32) {                                // wave 0: async DMA the A tile
    stage_a_tile_tdm(A, rowbase, lds_a, N);
    __builtin_amdgcn_s_wait_tensorcnt(0);
  }
  if (tid >= 32 && tid < 48) {
    const int t = tid - 32;
    s_nni[t] = nn_idx[rowbase + t];
    s_nns[t] = nn_src[rowbase + t];
  }
  __syncthreads();

  Frag16 af[8];
  load_a_frags((const unsigned int*)lds_a, m, hi, af);

  int growv[8], nni_r[8], nnact[8];
  float ssum[8], dcap[8];
#pragma unroll
  for (int r = 0; r < 8; ++r) {
    const int rl = r + hi * 8;
    growv[r] = rowbase + rl;
    nni_r[r] = s_nni[rl];
    nnact[r] = (s_nns[rl] == mask_src_val);
    ssum[r] = 0.f;
    dcap[r] = NEG_INF;
  }

  const int ntiles = N / 16;
  for (int nt = wv; nt < ntiles; nt += 8) {
    const int c = nt * 16 + m;                   // this lane's column
    if (nt + 8 < ntiles)
      __builtin_prefetch(B + (size_t)(c + 128) * DIM, 0, 1);
    v8f acc = tile_gemm(af, B, c, hi);
#pragma unroll
    for (int r = 0; r < 8; ++r) {
      float v = acc[r] * INV_T;
      const bool isdiag = (c == growv[r]);
      dcap[r] = isdiag ? v : dcap[r];            // positive-pair logit (pre-mask)
      const bool masked = (nnact[r] && c == nni_r[r]) || (mask_diag && isdiag);
      v = masked ? NEG_INF : v;
      ssum[r] += __expf(v - LSE_C);              // v_exp co-executes with WMMA
    }
  }

  // single cross-lane reduction at the end
#pragma unroll
  for (int r = 0; r < 8; ++r) {
    float s = ssum[r];
#pragma unroll
    for (int mask = 1; mask < 16; mask <<= 1) s += __shfl_xor(s, mask, 32);
    float d = dcap[r];
#pragma unroll
    for (int mask = 1; mask < 16; mask <<= 1) d = fmaxf(d, __shfl_xor(d, mask, 32));
    if (m == 0) { s_s[wv][r + hi * 8] = s; s_dw[wv][r + hi * 8] = d; }
  }
  __syncthreads();

  if (tid < 16) {
    float S = 0.f;
    for (int w = 0; w < 8; ++w) S += s_s[w][tid];
    m_out[rowbase + tid] = LSE_C;
    s_out[rowbase + tid] = S;
    if (dvec != nullptr) {
      float d = NEG_INF;
      for (int w = 0; w < 8; ++w) d = fmaxf(d, s_dw[w][tid]);
      dvec[rowbase + tid] = d;
    }
  }
}

// ---------------------------------------------------------------------------
// 5) Merge the ab/aa sum-exp halves, per-row loss term (already / N)
// ---------------------------------------------------------------------------
__global__ __launch_bounds__(256)
void combine_loss_kernel(const float* __restrict__ m_ab, const float* __restrict__ s_ab,
                         const float* __restrict__ m_aa, const float* __restrict__ s_aa,
                         const float* __restrict__ dvec, float* __restrict__ lossbuf, int N) {
  const int i = blockIdx.x * blockDim.x + threadIdx.x;
  if (i >= N) return;
  const float M = fmaxf(m_ab[i], m_aa[i]);
  const float S = s_ab[i] * __expf(m_ab[i] - M) + s_aa[i] * __expf(m_aa[i] - M);
  const float lse = M + __logf(S);
  lossbuf[i] = (lse - dvec[i]) * (1.0f / (float)N);
}

// 6) Deterministic fixed-order final reduction -> scalar loss
__global__ __launch_bounds__(256)
void final_sum_kernel(const float* __restrict__ lossbuf, float* __restrict__ out, int total) {
  __shared__ float red[256];
  const int tid = threadIdx.x;
  float a = 0.f;
  for (int k = tid; k < total; k += 256) a += lossbuf[k];
  red[tid] = a;
  __syncthreads();
  for (int s = 128; s > 0; s >>= 1) {
    if (tid < s) red[tid] += red[tid + s];
    __syncthreads();
  }
  if (tid == 0) out[0] = red[0];
}

extern "C" void kernel_launch(void* const* d_in, const int* in_sizes, int n_in,
                              void* d_out, int out_size, void* d_ws, size_t ws_size,
                              hipStream_t stream) {
  const float* zi = (const float*)d_in[0];
  const float* zj = (const float*)d_in[1];
  const int N = in_sizes[0] / DIM;               // 4096

  char* ws = (char*)d_ws;
  size_t off = 0;
  auto take = [&](size_t bytes) -> void* {
    void* p = ws + off;
    off += (bytes + 255) & ~(size_t)255;
    return p;
  };

  _Float16* zi_h  = (_Float16*)take((size_t)N * DIM * sizeof(_Float16));
  _Float16* zj_h  = (_Float16*)take((size_t)N * DIM * sizeof(_Float16));
  _Float16* znn_i = (_Float16*)take((size_t)N * DIM * sizeof(_Float16));
  _Float16* znn_j = (_Float16*)take((size_t)N * DIM * sizeof(_Float16));

  float* mv_aa_i = (float*)take(N * 4); int* mi_aa_i = (int*)take(N * 4);
  float* mv_ab_i = (float*)take(N * 4); int* mi_ab_i = (int*)take(N * 4);
  float* mv_aa_j = (float*)take(N * 4); int* mi_aa_j = (int*)take(N * 4);
  float* mv_ab_j = (float*)take(N * 4); int* mi_ab_j = (int*)take(N * 4);
  int* nni_i = (int*)take(N * 4); int* nns_i = (int*)take(N * 4);
  int* nni_j = (int*)take(N * 4); int* nns_j = (int*)take(N * 4);
  float* m_ab_i = (float*)take(N * 4); float* s_ab_i = (float*)take(N * 4);
  float* m_aa_i = (float*)take(N * 4); float* s_aa_i = (float*)take(N * 4);
  float* m_ab_j = (float*)take(N * 4); float* s_ab_j = (float*)take(N * 4);
  float* m_aa_j = (float*)take(N * 4); float* s_aa_j = (float*)take(N * 4);
  float* d_i = (float*)take(N * 4);    float* d_j = (float*)take(N * 4);
  float* lossbuf = (float*)take((size_t)2 * N * 4);

  normalize_f16_kernel<<<N, DIM, 0, stream>>>(zi, zi_h);
  normalize_f16_kernel<<<N, DIM, 0, stream>>>(zj, zj_h);

  dim3 gB(N / 16), b256(256);
  rowmax_kernel<<<gB, b256, 0, stream>>>(zi_h, zi_h, mv_aa_i, mi_aa_i, N);
  rowmax_kernel<<<gB, b256, 0, stream>>>(zi_h, zj_h, mv_ab_i, mi_ab_i, N);
  rowmax_kernel<<<gB, b256, 0, stream>>>(zj_h, zj_h, mv_aa_j, mi_aa_j, N);
  rowmax_kernel<<<gB, b256, 0, stream>>>(zj_h, zi_h, mv_ab_j, mi_ab_j, N);

  select_gather_kernel<<<N, 128, 0, stream>>>(mv_aa_i, mi_aa_i, mv_ab_i, mi_ab_i,
                                              zi_h, zj_h, znn_i, nni_i, nns_i);
  select_gather_kernel<<<N, 128, 0, stream>>>(mv_aa_j, mi_aa_j, mv_ab_j, mi_ab_j,
                                              zj_h, zi_h, znn_j, nni_j, nns_j);

  lse_kernel<<<gB, b256, 0, stream>>>(znn_i, zj_h, nni_i, nns_i, 1, 0, m_ab_i, s_ab_i, d_i, N);
  lse_kernel<<<gB, b256, 0, stream>>>(znn_i, zi_h, nni_i, nns_i, 0, 1, m_aa_i, s_aa_i, nullptr, N);
  lse_kernel<<<gB, b256, 0, stream>>>(znn_j, zi_h, nni_j, nns_j, 1, 0, m_ab_j, s_ab_j, d_j, N);
  lse_kernel<<<gB, b256, 0, stream>>>(znn_j, zj_h, nni_j, nns_j, 0, 1, m_aa_j, s_aa_j, nullptr, N);

  combine_loss_kernel<<<(N + 255) / 256, 256, 0, stream>>>(m_ab_i, s_ab_i, m_aa_i, s_aa_i,
                                                           d_i, lossbuf, N);
  combine_loss_kernel<<<(N + 255) / 256, 256, 0, stream>>>(m_ab_j, s_ab_j, m_aa_j, s_aa_j,
                                                           d_j, lossbuf + N, N);
  final_sum_kernel<<<1, 256, 0, stream>>>(lossbuf, (float*)d_out, 2 * N);
}
